// WeightDemodulation_12360915878619
// MI455X (gfx1250) — compile-verified, matrix operand order checked by reading, not run on hardware
//
#include <hip/hip_runtime.h>
#include <hip/hip_fp16.h>

typedef _Float16 h16;
typedef __attribute__((ext_vector_type(8)))  _Float16 v8h;
typedef __attribute__((ext_vector_type(16))) _Float16 v16h;
typedef __attribute__((ext_vector_type(8)))  float    v8f;

#define CIN   512
#define COUT  512
#define HH    64
#define WW    64
#define NB    16
#define COND  512
#define EPSC  1e-8f

// ---------------- prologue kernels (tiny) ----------------

// gamma[b,c] = y[b,:] . w_gamma[c,:] + b_gamma[c]        (16x512)
__global__ __launch_bounds__(256)
void k_gamma(const float* __restrict__ y, const float* __restrict__ wg,
             const float* __restrict__ bg, float* __restrict__ gamma) {
  int i = blockIdx.x * 256 + threadIdx.x;          // 0..8191
  int b = i >> 9, c = i & 511;
  const float* yb = y + (size_t)b * COND;
  const float* wr = wg + (size_t)c * COND;
  float s = bg[c];
  for (int k = 0; k < COND; ++k) s += yb[k] * wr[k];
  gamma[i] = s;
}

// sq[o,i] = sum_{kh,kw} w_conv[o,i,kh,kw]^2              (512x512)
__global__ __launch_bounds__(256)
void k_sq(const float* __restrict__ wconv, float* __restrict__ sq) {
  int i = blockIdx.x * 256 + threadIdx.x;          // o*512 + ci
  const float* wp = wconv + (size_t)i * 9;
  float s = 0.f;
#pragma unroll
  for (int p = 0; p < 9; ++p) s += wp[p] * wp[p];
  sq[i] = s;
}

// dsc[b,o] = rsqrt( sum_i gamma[b,i]^2 * sq[o,i] + eps )  (16x512)
__global__ __launch_bounds__(256)
void k_demod(const float* __restrict__ gamma, const float* __restrict__ sq,
             float* __restrict__ dsc) {
  int i = blockIdx.x * 256 + threadIdx.x;          // b*512 + o
  int b = i >> 9, o = i & 511;
  const float* gb = gamma + (size_t)b * CIN;
  const float* so = sq + (size_t)o * CIN;
  float s = EPSC;
  for (int c = 0; c < CIN; ++c) { float g = gb[c]; s += g * g * so[c]; }
  dsc[i] = rsqrtf(s);
}

// ---------------- main modulated-conv implicit-GEMM kernel ----------------
// grid: (HH/2, COUT/64, NB); block: 256 threads = 8 wave32
// WG tile: 64 cout x (2 rows x 64 cols); wave tile: 32 cout x 32 px (2x2 WMMA)

__global__ __launch_bounds__(256, 2)
void k_conv(const float* __restrict__ x, const float* __restrict__ wconv,
            const float* __restrict__ gamma, const float* __restrict__ dsc,
            float* __restrict__ out) {
  // [halo row 0..3][padded col 0..65][channel 0..31], channel contiguous so a
  // lane's 8-consecutive-K fragment chunk is one aligned 16B ds_load_b128.
  __shared__ __align__(16) h16 xs[4][66][32];
  __shared__ __align__(16) h16 wt[9][64][32];

  const int tid  = threadIdx.x;
  const int lane = tid & 31;
  const int wave = tid >> 5;
  const int h0     = blockIdx.x * 2;          // first output row of tile
  const int coBase = blockIdx.y * 64;         // cout tile base
  const int batch  = blockIdx.z;

  // Padding columns (image col -1 and 64) are always outside the image for a
  // full-width tile with pad=1: zero them once (bijective tid -> (r,side,c)).
  {
    int r = tid >> 6, cs = (tid >> 5) & 1, c = tid & 31;
    xs[r][cs ? 65 : 0][c] = (h16)0.f;
  }

  const int waveM = wave & 1;                 // cout sub-tile (32 each)
  const int waveN = wave >> 1;                // pixel sub-tile (32 each)

  const v8f vz = {0.f, 0.f, 0.f, 0.f, 0.f, 0.f, 0.f, 0.f};
  v8f acc[2][2] = {{vz, vz}, {vz, vz}};

  const float* xb = x + (size_t)batch * CIN * HH * WW;
  const float* gb = gamma + (size_t)batch * CIN;

  // ISA 16-bit A/B layout: lane l -> row/col = l&15; K chunks [kb,kb+8) and
  // [16+kb,16+kb+8) with kb = 8*(l>=16).
  const int mrow = lane & 15;
  const int kb   = (lane >> 4) << 3;

  for (int ci = 0; ci < CIN; ci += 32) {
    __syncthreads();

    // Stage modulated input rows h0-1 .. h0+2 (coalesced along w), f32->f16.
    {
      const int col = tid & 63;
      for (int slice = tid >> 6; slice < 4 * 32; slice += 4) {
        int r = slice >> 5, c = slice & 31;
        int hr = h0 - 1 + r;
        float v = 0.f;
        if (0 <= hr && hr < HH)
          v = xb[(size_t)(ci + c) * (HH * WW) + hr * WW + col] * gb[ci + c];
        xs[r][col + 1][c] = (h16)v;
      }
    }
    // Stage weight tile: all 9 taps for 64 couts x 32 cins, f32->f16.
    for (int idx = tid; idx < 64 * 32; idx += 256) {
      int o = idx >> 5, c = idx & 31;
      const float* wp = wconv + ((size_t)(coBase + o) * CIN + (ci + c)) * 9;
#pragma unroll
      for (int p = 0; p < 9; ++p) wt[p][o][c] = (h16)wp[p];
    }
    __syncthreads();

#pragma unroll
    for (int p = 0; p < 9; ++p) {
      const int kh = p / 3, kw = p % 3;       // constants after unroll
      v16h a[2], bfr[2];
#pragma unroll
      for (int mt = 0; mt < 2; ++mt) {
        const h16* ap = &wt[p][waveM * 32 + mt * 16 + mrow][0];
        v8h lo = *reinterpret_cast<const v8h*>(ap + kb);
        v8h hi = *reinterpret_cast<const v8h*>(ap + 16 + kb);
        a[mt] = __builtin_shufflevector(lo, hi, 0, 1, 2, 3, 4, 5, 6, 7,
                                        8, 9, 10, 11, 12, 13, 14, 15);
      }
#pragma unroll
      for (int nt = 0; nt < 2; ++nt) {
        int pix = waveN * 32 + nt * 16 + mrow;      // 0..127 within tile
        int hh = pix >> 6, wc = pix & 63;
        // input row = h0+hh+kh-1 -> halo row hh+kh; input col = wc+kw-1 -> padded col wc+kw
        const h16* bp = &xs[hh + kh][wc + kw][0];
        v8h lo = *reinterpret_cast<const v8h*>(bp + kb);
        v8h hi = *reinterpret_cast<const v8h*>(bp + 16 + kb);
        bfr[nt] = __builtin_shufflevector(lo, hi, 0, 1, 2, 3, 4, 5, 6, 7,
                                          8, 9, 10, 11, 12, 13, 14, 15);
      }
#pragma unroll
      for (int mt = 0; mt < 2; ++mt)
#pragma unroll
        for (int nt = 0; nt < 2; ++nt)
          acc[mt][nt] = __builtin_amdgcn_wmma_f32_16x16x32_f16(
              false, a[mt], false, bfr[nt], (short)0, acc[mt][nt], false, false);
    }
  }

  // Epilogue: D layout is VGPR v -> M = v + 8*(lane>=16), N = lane&15.
  const int nlo  = lane & 15;
  const int mofs = (lane >> 4) << 3;
  float* ob = out + (size_t)batch * COUT * HH * WW;
  const float* db = dsc + (size_t)batch * COUT;
#pragma unroll
  for (int mt = 0; mt < 2; ++mt) {
    int coB = coBase + waveM * 32 + mt * 16 + mofs;
#pragma unroll
    for (int nt = 0; nt < 2; ++nt) {
      int pix = waveN * 32 + nt * 16 + nlo;
      int hh = pix >> 6, wc = pix & 63;
      size_t o0 = ((size_t)coB * HH + (h0 + hh)) * WW + wc;
#pragma unroll
      for (int v = 0; v < 8; ++v)
        ob[o0 + (size_t)v * (HH * WW)] = acc[mt][nt][v] * db[coB + v];
    }
  }
}

// ---------------- launcher ----------------

extern "C" void kernel_launch(void* const* d_in, const int* in_sizes, int n_in,
                              void* d_out, int out_size, void* d_ws, size_t ws_size,
                              hipStream_t stream) {
  (void)in_sizes; (void)n_in; (void)out_size; (void)ws_size;
  const float* x      = (const float*)d_in[0];   // [16,512,64,64]
  const float* y      = (const float*)d_in[1];   // [16,512]
  const float* wconv  = (const float*)d_in[2];   // [512,512,3,3]
  const float* wgamma = (const float*)d_in[3];   // [512,512]
  const float* bgamma = (const float*)d_in[4];   // [512]
  float* out = (float*)d_out;                    // [16,512,64,64]

  // workspace: gamma (8192 f32) | dsc (8192 f32) | sq (262144 f32)  ~1.1 MB
  float* gamma = (float*)d_ws;
  float* dsc   = gamma + NB * CIN;
  float* sq    = dsc + NB * COUT;

  k_gamma<<<(NB * CIN) / 256, 256, 0, stream>>>(y, wgamma, bgamma, gamma);
  k_sq<<<(COUT * CIN) / 256, 256, 0, stream>>>(wconv, sq);
  k_demod<<<(NB * COUT) / 256, 256, 0, stream>>>(gamma, sq, dsc);

  dim3 grid(HH / 2, COUT / 64, NB);
  k_conv<<<grid, 256, 0, stream>>>(x, wconv, gamma, dsc, out);
}
